// EFIN_32332513804349
// MI455X (gfx1250) — compile-verified, matrix operand order checked by reading, not run on hardware
//
#include <hip/hip_runtime.h>

#define B_SZ   8192
#define F_DIM  100
#define F_PAD  112
#define H_DIM  128
#define HC_DIM 512
#define FH     12800   // F_DIM * H_DIM

typedef _Float16 v16h __attribute__((ext_vector_type(16)));
typedef _Float16 h8   __attribute__((ext_vector_type(8)));
typedef float    v8f  __attribute__((ext_vector_type(8)));

__device__ __forceinline__ float sigf(float x) { return 1.f / (1.f + __expf(-x)); }
__device__ __forceinline__ float eluf(float x) { return x > 0.f ? x : (__expf(x) - 1.f); }

// A-operand fragment: 16x32 f16 tile from row-major [*, ld] f16 matrix.
// Lane layout per CDNA5 ISA 7.12.2 (16-bit A 16x32): lanes 0-15 hold K 0-7,16-23;
// lanes 16-31 hold K 8-15,24-31. Two contiguous 16B LDS reads per lane.
__device__ __forceinline__ v16h frag_a16(const _Float16* base, int ld, int r0, int k0) {
  const int lane = threadIdx.x & 31;
  const int half = lane >> 4;
  const _Float16* p = base + (r0 + (lane & 15)) * ld + k0 + half * 8;
  h8 lo = *(const h8*)p;
  h8 hi = *(const h8*)(p + 16);
  return __builtin_shufflevector(lo, hi, 0,1,2,3,4,5,6,7,8,9,10,11,12,13,14,15);
}

// B-operand fragment from row-major [K][N] f16 matrix: lane holds column n0+(lane&15),
// K interleave mirrors the A layout (B fragment == A-layout of B^T).
__device__ __forceinline__ v16h frag_b16(const _Float16* base, int ld, int k0, int n0) {
  const int lane = threadIdx.x & 31;
  const int half = lane >> 4;
  const _Float16* col = base + n0 + (lane & 15);
  v16h b;
#pragma unroll
  for (int i = 0; i < 16; ++i) {
    const int k = k0 + ((i >> 3) << 4) + half * 8 + (i & 7);
    b[i] = col[k * ld];
  }
  return b;
}

// B-operand fragment from row-major [K][N] f32 global weight, converted to f16.
__device__ __forceinline__ v16h frag_b_f32(const float* W, int ld, int k0, int n0) {
  const int lane = threadIdx.x & 31;
  const int half = lane >> 4;
  const float* col = W + n0 + (lane & 15);
  v16h b;
#pragma unroll
  for (int i = 0; i < 16; ++i) {
    const int k = k0 + ((i >> 3) << 4) + half * 8 + (i & 7);
    b[i] = (_Float16)col[k * ld];
  }
  return b;
}

#define WMMA_F16(a, b, c) \
  __builtin_amdgcn_wmma_f32_16x16x32_f16(false, (a), false, (b), (short)0, (c), false, false)

// Async DMA: 16 bytes global -> LDS, tracked by ASYNCcnt (no VGPR roundtrip).
// GVS addressing: mem = saddr + vaddr32; LDS dest = low 32 bits of generic ptr.
__device__ __forceinline__ void async_ld_b128(unsigned lds_addr, unsigned goff,
                                              const void* base) {
  asm volatile("global_load_async_to_lds_b128 %0, %1, %2"
               :: "v"(lds_addr), "v"(goff), "s"(base) : "memory");
}
__device__ __forceinline__ unsigned lds_lo32(const void* p) {
  return (unsigned)(unsigned long long)p;
}

// ---------------------------------------------------------------------------
// Kernel A: one workgroup (256 threads = 8 waves) per batch row.
// Fuses: x_rep, feature-axis norm, QKV GEMMs, sigmoid+softmax attention,
// xx = attw@V (to global f16), y = x_rep@a2w, interaction attention, xt.
// ~146 KB LDS, matrix work in v_wmma_f32_16x16x32_f16.
// ---------------------------------------------------------------------------
__global__ __launch_bounds__(256) void efin_attn(
    const float* __restrict__ fl, const float* __restrict__ emb,
    const float* __restrict__ t_w, const float* __restrict__ t_b,
    const float* __restrict__ a1w,
    const float* __restrict__ qw, const float* __restrict__ qb,
    const float* __restrict__ kw, const float* __restrict__ kb,
    const float* __restrict__ vw, const float* __restrict__ vb,
    const float* __restrict__ a2w, const float* __restrict__ a2b,
    const float* __restrict__ a3w,
    _Float16* __restrict__ xx_out, float* __restrict__ xt_out)
{
  __shared__ _Float16 s_xrep[F_PAD][H_DIM];
  __shared__ _Float16 s_nx[F_PAD][H_DIM];   // reused as attention weights
  __shared__ _Float16 s_Q[F_PAD][H_DIM];
  __shared__ _Float16 s_K[F_PAD][H_DIM];    // reused as y = x_rep@a2w + a2b
  __shared__ _Float16 s_V[F_PAD][H_DIM];
  __shared__ float s_fl[F_PAD];
  __shared__ float s_rn[H_DIM];
  __shared__ float s_red[2][H_DIM];
  __shared__ float s_s1[H_DIM];
  __shared__ float s_attn[F_PAD];
  __shared__ float s_mx[2];

  const int t    = threadIdx.x;
  const int lane = t & 31;
  const int wave = t >> 5;
  const int b    = blockIdx.x;

  // phase 0: features + s1 = sigmoid((t_w+t_b) @ a1w)  (same for all b)
  if (t < F_PAD) s_fl[t] = (t < F_DIM) ? fl[b * F_DIM + t] : 0.f;
  if (t < H_DIM) {
    float acc = 0.f;
    for (int h = 0; h < H_DIM; ++h) acc += (t_w[h] + t_b[h]) * a1w[h * H_DIM + t];
    s_s1[t] = sigf(acc);
  }
  __syncthreads();

  // phase 1: x_rep = fl * emb; norm over feature axis per h; nx = x_rep / norm
  {
    const int h = t & (H_DIM - 1);
    const int fh = t >> 7;
    float nsq = 0.f;
    for (int f = fh; f < F_PAD; f += 2) {
      const float e = (f < F_DIM) ? emb[f * H_DIM + h] : 0.f;
      const float x = s_fl[f] * e;
      s_xrep[f][h] = (_Float16)x;
      nsq += x * x;
    }
    s_red[fh][h] = nsq;
  }
  __syncthreads();
  if (t < H_DIM) {
    const float ns = s_red[0][t] + s_red[1][t];
    s_rn[t] = ns > 0.f ? rsqrtf(ns) : 0.f;
  }
  __syncthreads();
  {
    const int h = t & (H_DIM - 1);
    const int fh = t >> 7;
    const float rn = s_rn[h];
    for (int f = fh; f < F_PAD; f += 2)
      s_nx[f][h] = (_Float16)((float)s_xrep[f][h] * rn);
  }
  __syncthreads();

  // phase 2: Q,K,V = nx @ W + bias (WMMA, 7x8 tiles each, K=128).
  // Runtime ternary selection (no const LDS-pointer aggregates: those lower to
  // addrspacecast static initializers that ld.lld rejects).
  for (int m = 0; m < 3; ++m) {
    const float* W    = (m == 0) ? qw : (m == 1) ? kw : vw;
    const float* bias = (m == 0) ? qb : (m == 1) ? kb : vb;
    _Float16 (*dst)[H_DIM] = (m == 0) ? s_Q : (m == 1) ? s_K : s_V;
    for (int tile = wave; tile < 56; tile += 8) {
      const int fi = tile >> 3, hj = tile & 7;
      v8f acc = {};
#pragma unroll
      for (int kk = 0; kk < 4; ++kk) {
        v16h a  = frag_a16(&s_nx[0][0], H_DIM, fi * 16, kk * 32);
        v16h bb = frag_b_f32(W, H_DIM, kk * 32, hj * 16);
        acc = WMMA_F16(a, bb, acc);
      }
      const int r0 = fi * 16 + (lane >> 4) * 8;
      const int c  = hj * 16 + (lane & 15);
      const float bv = bias[c];
#pragma unroll
      for (int j = 0; j < 8; ++j) dst[r0 + j][c] = (_Float16)(acc[j] + bv);
    }
  }
  __syncthreads();

  // phase 3: att = sigmoid(Q K^T / sqrt(H)) into s_nx (reused);
  // B-fragment of K^T == A-layout fragment of K.
  _Float16 (*s_att)[H_DIM] = s_nx;
  {
    const float scale = 0.08838834764831845f; // 1/sqrt(128)
    for (int tile = wave; tile < 49; tile += 8) {
      const int fi = tile / 7, gj = tile % 7;
      v8f acc = {};
#pragma unroll
      for (int kk = 0; kk < 4; ++kk) {
        v16h a  = frag_a16(&s_Q[0][0], H_DIM, fi * 16, kk * 32);
        v16h bb = frag_a16(&s_K[0][0], H_DIM, gj * 16, kk * 32);
        acc = WMMA_F16(a, bb, acc);
      }
      const int r0 = fi * 16 + (lane >> 4) * 8;
      const int c  = gj * 16 + (lane & 15);
#pragma unroll
      for (int j = 0; j < 8; ++j)
        s_att[r0 + j][c] = (_Float16)sigf(acc[j] * scale);
    }
    for (int idx = t; idx < F_PAD * 16; idx += 256)   // zero K-pad cols 112..127
      s_att[idx >> 4][112 + (idx & 15)] = (_Float16)0.f;
  }
  __syncthreads();

  // phase 4: row softmax over the 100 valid cols (inputs in (0,1): no max shift)
  if (t < F_PAD) {
    float sum = 0.f;
    for (int g = 0; g < F_DIM; ++g) sum += __expf((float)s_att[t][g]);
    const float rs = 1.f / sum;
    for (int g = 0; g < F_DIM; ++g)
      s_att[t][g] = (_Float16)(__expf((float)s_att[t][g]) * rs);
    for (int g = F_DIM; g < F_PAD; ++g) s_att[t][g] = (_Float16)0.f;
  }
  __syncthreads();

  // phase 5: xx = attw @ V -> global f16 [B, F*H]
  for (int tile = wave; tile < 56; tile += 8) {
    const int fi = tile >> 3, hj = tile & 7;
    v8f acc = {};
#pragma unroll
    for (int kk = 0; kk < 4; ++kk) {
      v16h a  = frag_a16(&s_att[0][0], H_DIM, fi * 16, kk * 32);
      v16h bb = frag_b16(&s_V[0][0], H_DIM, kk * 32, hj * 16);
      acc = WMMA_F16(a, bb, acc);
    }
    const int r0 = fi * 16 + (lane >> 4) * 8;
    const int c  = hj * 16 + (lane & 15);
#pragma unroll
    for (int j = 0; j < 8; ++j) {
      const int f = r0 + j;
      if (f < F_DIM) xx_out[(size_t)b * FH + f * H_DIM + c] = (_Float16)acc[j];
    }
  }

  // phase 6: y = x_rep @ a2w + a2b -> s_K (reused); disjoint from phase-5 buffers
  _Float16 (*s_y)[H_DIM] = s_K;
  for (int tile = wave; tile < 56; tile += 8) {
    const int fi = tile >> 3, hj = tile & 7;
    v8f acc = {};
#pragma unroll
    for (int kk = 0; kk < 4; ++kk) {
      v16h a  = frag_a16(&s_xrep[0][0], H_DIM, fi * 16, kk * 32);
      v16h bb = frag_b_f32(a2w, H_DIM, kk * 32, hj * 16);
      acc = WMMA_F16(a, bb, acc);
    }
    const int r0 = fi * 16 + (lane >> 4) * 8;
    const int c  = hj * 16 + (lane & 15);
    const float bv = a2b[c];
#pragma unroll
    for (int j = 0; j < 8; ++j) s_y[r0 + j][c] = (_Float16)(acc[j] + bv);
  }
  __syncthreads();

  // phase 7: interaction attention over features
  if (t < F_DIM) {
    float acc = 0.f;
    for (int h = 0; h < H_DIM; ++h) {
      float r = s_s1[h] + sigf((float)s_y[t][h]);
      r = r > 0.f ? r : 0.f;
      acc += r * a3w[h];
    }
    s_attn[t] = acc;
  }
  __syncthreads();
  if (t == 0) {
    float mx = -1e30f;
    for (int f = 0; f < F_DIM; ++f) mx = fmaxf(mx, s_attn[f]);
    float sum = 0.f;
    for (int f = 0; f < F_DIM; ++f) sum += __expf(s_attn[f] - mx);
    s_mx[0] = mx; s_mx[1] = 1.f / sum;
  }
  __syncthreads();
  if (t < F_DIM) s_attn[t] = __expf(s_attn[t] - s_mx[0]) * s_mx[1];
  __syncthreads();

  // phase 8: xt = attn @ x_rep -> global f32 [B, H]
  if (t < H_DIM) {
    float acc = 0.f;
    for (int f = 0; f < F_DIM; ++f) acc += s_attn[f] * (float)s_xrep[f][t];
    xt_out[(size_t)b * H_DIM + t] = acc;
  }
}

// ---------------------------------------------------------------------------
// c1w f32 -> f16 conversion (one pass per launch; deterministic)
// ---------------------------------------------------------------------------
__global__ void efin_cvt(const float* __restrict__ src, _Float16* __restrict__ dst, int n) {
  const int i = blockIdx.x * blockDim.x + threadIdx.x;
  if (i < n) dst[i] = (_Float16)src[i];
}

// ---------------------------------------------------------------------------
// Kernel B: c1 GEMM  [8192,12800] @ [12800,512] + bias, ELU -> c_act f16.
// 128x128 tile per workgroup; K streamed in 32-wide stages via
// global_load_async_to_lds_b128 (ASYNCcnt DMA) with double-buffered LDS:
// stage k+1 issued into the alternate buffer, s_wait_asynccnt 0x4 guarantees
// stage k (the 4 oldest per-thread ops, in-order completion) has landed.
// ---------------------------------------------------------------------------
__global__ __launch_bounds__(256) void efin_c1(
    const _Float16* __restrict__ xx, const _Float16* __restrict__ w16,
    const float* __restrict__ c1b, _Float16* __restrict__ cact)
{
  __shared__ __align__(16) _Float16 As[2][128][40];   // 32 + 8 pad halfs
  __shared__ __align__(16) _Float16 Bs[2][32][136];   // 128 + 8 pad halfs
  const int t = threadIdx.x, lane = t & 31, wave = t >> 5;
  const int m0 = (blockIdx.x >> 2) * 128;
  const int n0 = (blockIdx.x & 3) * 128;

  // per-thread staging coordinates (16 halfs = 2 x b128 each for A and B)
  const int arow = t >> 1, aseg = (t & 1) * 16;
  const int brow = t >> 3, bseg = (t & 7) * 16;
  const unsigned ga0 = (unsigned)((m0 + arow) * FH + aseg) * 2u;   // + k0*2
  const unsigned gb0 = (unsigned)(brow * HC_DIM + n0 + bseg) * 2u; // + k0*1024

  v8f acc[8];
#pragma unroll
  for (int i = 0; i < 8; ++i) acc[i] = (v8f){};

  // prologue: stage 0 into buffer 0
  {
    const unsigned la = lds_lo32(&As[0][arow][aseg]);
    const unsigned lb = lds_lo32(&Bs[0][brow][bseg]);
    async_ld_b128(la,      ga0,      xx);
    async_ld_b128(la + 16, ga0 + 32, xx);
    async_ld_b128(lb,      gb0,      w16);
    async_ld_b128(lb + 16, gb0 + 32, w16);
  }

  const int NIT = FH / 32;   // 400
  for (int it = 0; it < NIT; ++it) {
    const int cur = it & 1;
    if (it + 1 < NIT) {
      const unsigned k2 = (unsigned)(it + 1) * 32u;
      const unsigned la = lds_lo32(&As[cur ^ 1][arow][aseg]);
      const unsigned lb = lds_lo32(&Bs[cur ^ 1][brow][bseg]);
      async_ld_b128(la,      ga0 + k2 * 2u,        xx);
      async_ld_b128(la + 16, ga0 + k2 * 2u + 32,   xx);
      async_ld_b128(lb,      gb0 + k2 * 1024u,     w16);
      async_ld_b128(lb + 16, gb0 + k2 * 1024u + 32, w16);
      asm volatile("s_wait_asynccnt 0x4" ::: "memory");
    } else {
      asm volatile("s_wait_asynccnt 0x0" ::: "memory");
    }
    __syncthreads();                       // stage `it` visible to all waves
    v16h a = frag_a16(&As[cur][0][0], 40, wave * 16, 0);
#pragma unroll
    for (int ni = 0; ni < 8; ++ni) {
      v16h bb = frag_b16(&Bs[cur][0][0], 136, 0, ni * 16);
      acc[ni] = WMMA_F16(a, bb, acc[ni]);
    }
    __syncthreads();                       // done reading buf `cur`
  }

  const int r0 = wave * 16 + (lane >> 4) * 8;
#pragma unroll
  for (int ni = 0; ni < 8; ++ni) {
    const int c = n0 + ni * 16 + (lane & 15);
    const float bv = c1b[c];
#pragma unroll
    for (int j = 0; j < 8; ++j) {
      const float v = eluf(acc[ni][j] + bv);
      cact[(size_t)(m0 + r0 + j) * HC_DIM + c] = (_Float16)v;
    }
  }
}

// ---------------------------------------------------------------------------
// Small MLP towers (VALU; <3% of total FLOPs)
// ---------------------------------------------------------------------------
template <int LD>
__device__ __forceinline__ void tower_layer(const float (*in)[LD], float (*outp)[LD],
    const float* __restrict__ W, const float* __restrict__ bias,
    int M, int K, int N, int t) {
  for (int idx = t; idx < M * N; idx += 256) {
    const int r = idx / N, n = idx - r * N;
    float acc = bias[n];
    for (int k = 0; k < K; ++k) acc += in[r][k] * W[k * N + n];
    outp[r][n] = eluf(acc);
  }
  __syncthreads();
}

__global__ __launch_bounds__(256) void efin_ctower(
    const _Float16* __restrict__ cact,
    const float* __restrict__ c2w, const float* __restrict__ c2b,
    const float* __restrict__ c3w, const float* __restrict__ c3b,
    const float* __restrict__ c4w, const float* __restrict__ c4b,
    const float* __restrict__ c5w, const float* __restrict__ c5b,
    const float* __restrict__ clw, const float* __restrict__ clb,
    const float* __restrict__ ctw, const float* __restrict__ ctb,
    float* __restrict__ out)
{
  __shared__ float a0[16][HC_DIM];
  __shared__ float a1[16][HC_DIM];
  const int t = threadIdx.x;
  const int m0 = blockIdx.x * 16;
  for (int idx = t; idx < 16 * HC_DIM; idx += 256)
    a0[idx >> 9][idx & 511] = (float)cact[(size_t)(m0 + (idx >> 9)) * HC_DIM + (idx & 511)];
  __syncthreads();
  tower_layer<HC_DIM>(a0, a1, c2w, c2b, 16, 512, 512, t);
  tower_layer<HC_DIM>(a1, a0, c3w, c3b, 16, 512, 256, t);
  tower_layer<HC_DIM>(a0, a1, c4w, c4b, 16, 256, 128, t);
  tower_layer<HC_DIM>(a1, a0, c5w, c5b, 16, 128, 64, t);
  if (t < 16) {
    const int r = t;
    float lg = clb[0], tau = ctb[0];
    for (int k = 0; k < 64; ++k) { lg += a0[r][k] * clw[k]; tau += a0[r][k] * ctw[k]; }
    out[m0 + r]            = lg;          // c_logit
    out[B_SZ + m0 + r]     = sigf(lg);    // c_prob
    out[2 * B_SZ + m0 + r] = tau;         // c_tau
  }
}

__global__ __launch_bounds__(256) void efin_utower(
    const float* __restrict__ xt,
    const float* __restrict__ u1w, const float* __restrict__ u1b,
    const float* __restrict__ u2w, const float* __restrict__ u2b,
    const float* __restrict__ u3w, const float* __restrict__ u3b,
    const float* __restrict__ u4w, const float* __restrict__ u4b,
    const float* __restrict__ tlw, const float* __restrict__ tlb,
    const float* __restrict__ utw, const float* __restrict__ utb,
    float* __restrict__ out)
{
  __shared__ float a0[64][H_DIM];
  __shared__ float a1[64][H_DIM];
  const int t = threadIdx.x;
  const int m0 = blockIdx.x * 64;
  for (int idx = t; idx < 64 * H_DIM; idx += 256)
    a0[idx >> 7][idx & 127] = xt[(size_t)(m0 + (idx >> 7)) * H_DIM + (idx & 127)];
  __syncthreads();
  tower_layer<H_DIM>(a0, a1, u1w, u1b, 64, 128, 128, t);
  tower_layer<H_DIM>(a1, a0, u2w, u2b, 64, 128, 64, t);
  tower_layer<H_DIM>(a0, a1, u3w, u3b, 64, 64, 32, t);
  tower_layer<H_DIM>(a1, a0, u4w, u4b, 64, 32, 16, t);
  if (t < 64) {
    const int r = t;
    float lg = tlb[0], tau = utb[0];
    for (int k = 0; k < 16; ++k) { lg += a0[r][k] * tlw[k]; tau += a0[r][k] * utw[k]; }
    out[3 * B_SZ + m0 + r] = lg;          // t_logit
    out[4 * B_SZ + m0 + r] = sigf(lg);    // t_prob
    out[5 * B_SZ + m0 + r] = tau;         // u_tau
  }
}

extern "C" void kernel_launch(void* const* d_in, const int* in_sizes, int n_in,
                              void* d_out, int out_size, void* d_ws, size_t ws_size,
                              hipStream_t stream) {
  const float* fl  = (const float*)d_in[0];
  const float* emb = (const float*)d_in[2];
  const float* t_w = (const float*)d_in[3];
  const float* t_b = (const float*)d_in[4];
  const float* qw  = (const float*)d_in[5];
  const float* qb  = (const float*)d_in[6];
  const float* kw  = (const float*)d_in[7];
  const float* kb  = (const float*)d_in[8];
  const float* vw  = (const float*)d_in[9];
  const float* vb  = (const float*)d_in[10];
  const float* a1w = (const float*)d_in[11];
  const float* a2w = (const float*)d_in[12];
  const float* a2b = (const float*)d_in[13];
  const float* a3w = (const float*)d_in[14];
  const float* c1w = (const float*)d_in[15];
  const float* c1b = (const float*)d_in[16];
  const float* c2w = (const float*)d_in[17];
  const float* c2b = (const float*)d_in[18];
  const float* c3w = (const float*)d_in[19];
  const float* c3b = (const float*)d_in[20];
  const float* c4w = (const float*)d_in[21];
  const float* c4b = (const float*)d_in[22];
  const float* c5w = (const float*)d_in[23];
  const float* c5b = (const float*)d_in[24];
  const float* clw = (const float*)d_in[25];
  const float* clb = (const float*)d_in[26];
  const float* ctw = (const float*)d_in[27];
  const float* ctb = (const float*)d_in[28];
  const float* u1w = (const float*)d_in[29];
  const float* u1b = (const float*)d_in[30];
  const float* u2w = (const float*)d_in[31];
  const float* u2b = (const float*)d_in[32];
  const float* u3w = (const float*)d_in[33];
  const float* u3b = (const float*)d_in[34];
  const float* u4w = (const float*)d_in[35];
  const float* u4b = (const float*)d_in[36];
  const float* tlw = (const float*)d_in[37];
  const float* tlb = (const float*)d_in[38];
  const float* utw = (const float*)d_in[39];
  const float* utb = (const float*)d_in[40];
  float* out = (float*)d_out;

  // workspace layout (~225 MB)
  _Float16* xx   = (_Float16*)d_ws;                       // [B, 12800] f16
  _Float16* w16  = xx + (size_t)B_SZ * FH;                // [12800, 512] f16
  float*    xt   = (float*)(w16 + (size_t)FH * HC_DIM);   // [B, 128] f32
  _Float16* cact = (_Float16*)(xt + (size_t)B_SZ * H_DIM);// [B, 512] f16

  efin_cvt<<<(FH * HC_DIM + 255) / 256, 256, 0, stream>>>(c1w, w16, FH * HC_DIM);
  efin_attn<<<B_SZ, 256, 0, stream>>>(fl, emb, t_w, t_b, a1w, qw, qb, kw, kb, vw, vb,
                                      a2w, a2b, a3w, xx, xt);
  efin_c1<<<256, 256, 0, stream>>>(xx, w16, c1b, cact);
  efin_ctower<<<B_SZ / 16, 256, 0, stream>>>(cact, c2w, c2b, c3w, c3b, c4w, c4b,
                                             c5w, c5b, clw, clb, ctw, ctb, out);
  efin_utower<<<B_SZ / 64, 256, 0, stream>>>(xt, u1w, u1b, u2w, u2b, u3w, u3b,
                                             u4w, u4b, tlw, tlb, utw, utb, out);
  (void)in_sizes; (void)n_in; (void)out_size; (void)ws_size;
}
